// CustomGATLayer_35021163331731
// MI455X (gfx1250) — matrix operand order repaired and checked by reading, not compiled, not run on hardware
//
#include <hip/hip_runtime.h>

#define N_NODES 8192
#define FDIM 256
#define ALPHA 0.2f

// ---- WMMA vector types (CDNA5 wave32 WMMA 16x16x32 bf16) ----
typedef __attribute__((ext_vector_type(16))) __bf16 bf16x16;
typedef __attribute__((ext_vector_type(8)))  float  f32x8;

union V16 { bf16x16 v; unsigned d[8]; };
union F4  { float4 v; float f[4]; };

__device__ __forceinline__ unsigned short f2bf(float f) {
  unsigned u = __float_as_uint(f);
  u += 0x7FFFu + ((u >> 16) & 1u);   // round to nearest even
  return (unsigned short)(u >> 16);
}

__device__ __forceinline__ float wave_max(float v) {
  #pragma unroll
  for (int s = 16; s > 0; s >>= 1) v = fmaxf(v, __shfl_xor(v, s, 32));
  return v;
}
__device__ __forceinline__ float wave_sum(float v) {
  #pragma unroll
  for (int s = 16; s > 0; s >>= 1) v += __shfl_xor(v, s, 32);
  return v;
}

// ============================================================
// Kernel 1: Wh = x @ W (f32), bf16 copy of Wh, e1 = Wh@a1, e2 = Wh@a2
// ============================================================
__global__ __launch_bounds__(256) void gat_wh_kernel(
    const float* __restrict__ x, const float* __restrict__ W,
    const float* __restrict__ a, unsigned short* __restrict__ whbf,
    float* __restrict__ e1, float* __restrict__ e2) {
  __shared__ float xs[FDIM];
  __shared__ float rbuf[256];
  const int i = blockIdx.x;
  const int t = threadIdx.x;

  xs[t] = x[(size_t)i * FDIM + t];
  __syncthreads();

  float acc = 0.f;
  #pragma unroll 8
  for (int k = 0; k < FDIM; ++k) acc = fmaf(xs[k], W[k * FDIM + t], acc);

  whbf[(size_t)i * FDIM + t] = f2bf(acc);

  const float p1 = acc * a[t];
  const float p2 = acc * a[FDIM + t];

  rbuf[t] = p1; __syncthreads();
  for (int s = 128; s > 0; s >>= 1) {
    if (t < s) rbuf[t] += rbuf[t + s];
    __syncthreads();
  }
  if (t == 0) e1[i] = rbuf[0];
  __syncthreads();
  rbuf[t] = p2; __syncthreads();
  for (int s = 128; s > 0; s >>= 1) {
    if (t < s) rbuf[t] += rbuf[t + s];
    __syncthreads();
  }
  if (t == 0) e2[i] = rbuf[0];
}

// ============================================================
// Kernel 1b: pack-pair transpose  WhT[col][jp] = {Wh[2jp+1][col], Wh[2jp][col]}
// grid 256 = 16 col-groups x 16 jp-groups; each thread: 1 col x 16 jp
// ============================================================
__global__ __launch_bounds__(256) void gat_tr_kernel(
    const unsigned short* __restrict__ whbf, unsigned* __restrict__ WhT) {
  const int t  = threadIdx.x;
  const int cg = blockIdx.x & 15, jg = blockIdx.x >> 4;
  const int col = cg * 16 + (t >> 4);
  const int jpb = jg * 256 + (t & 15) * 16;
  unsigned buf[16];
  #pragma unroll
  for (int p = 0; p < 16; ++p) {
    const int jp = jpb + p;
    const unsigned lo = whbf[(size_t)(2 * jp) * FDIM + col];      // L2-resident
    const unsigned hi = whbf[(size_t)(2 * jp + 1) * FDIM + col];
    buf[p] = lo | (hi << 16);
  }
  uint4* dst = (uint4*)(WhT + (size_t)col * (N_NODES / 2) + jpb);
  #pragma unroll
  for (int p = 0; p < 4; ++p)
    dst[p] = make_uint4(buf[4*p], buf[4*p+1], buf[4*p+2], buf[4*p+3]);
}

// ============================================================
// Kernel 2: fused masked softmax + attention @ Wh
// one block = 32 output rows, 256 threads = 8 waves
// K in 4 chunks of 2048; B tiles DMA'd via global_load_async_to_lds_b128
// ============================================================
#define MROWS  32
#define NCHUNK 4
#define CHUNK  2048
#define PSTR   2056                                // PSTR/2 % 64 == 4 -> conflict-free A gather
#define BSTRB  72                                  // col*36 mod 64 distinct; col*144 16B-aligned
#define LDS_P_BYTES (MROWS * PSTR * 2)             // 131584
#define LDS_B_OFF   LDS_P_BYTES
#define LDS_B_ONE   (FDIM * BSTRB * 2)             // 36864 (64 K-values per col)
#define LDS_R2_OFF  (LDS_B_OFF + 2 * LDS_B_ONE)    // 205312
#define LDS_MX_OFF  (LDS_R2_OFF + MROWS * 8 * 4)   // 206336
#define LDS_INV_OFF (LDS_MX_OFF + MROWS * 4)       // 206464
#define LDS_TOTAL   (LDS_INV_OFF + MROWS * 4)      // 206592 <= 320KB

// issue one 64-K B tile: per thread (column t) 8 x async B128 (16B runs)
__device__ __forceinline__ void issue_tile_async(const unsigned* __restrict__ WhT,
                                                 int t, int k0, int c, unsigned bs_base) {
  const int g0 = k0 >> 3;
  #pragma unroll
  for (int gl = 0; gl < 8; ++gl) {
    const unsigned* gsrc = WhT + (size_t)t * (N_NODES / 2) + (size_t)(16 * (g0 + gl) + 4 * c);
    const unsigned  ldst = bs_base + t * (BSTRB * 2) + gl * 16;
    asm volatile("global_load_async_to_lds_b128 %0, %1, off"
                 :: "v"(ldst), "v"((unsigned long long)(size_t)gsrc)
                 : "memory");
  }
}

__device__ __forceinline__ void mma_step(const unsigned short* Pl, const unsigned short* BsK,
                                         int k0, int m, int half, int w,
                                         f32x8& c00, f32x8& c01, f32x8& c10, f32x8& c11) {
  V16 A0, A1, B0, B1;
  const unsigned short* Pr0 = Pl + m * PSTR + k0;
  const unsigned short* Pr1 = Pl + (m + 16) * PSTR + k0;
  #pragma unroll
  for (int v = 0; v < 4; ++v) {
    const int K = half * 8 + 2 * v;
    A0.d[v] = *(const unsigned*)(Pr0 + K);
    A1.d[v] = *(const unsigned*)(Pr1 + K);
  }
  #pragma unroll
  for (int v = 0; v < 4; ++v) {
    const int K = 16 + half * 8 + 2 * v;
    A0.d[4 + v] = *(const unsigned*)(Pr0 + K);
    A1.d[4 + v] = *(const unsigned*)(Pr1 + K);
  }
  const int cb = w * 32;
  #pragma unroll
  for (int v = 0; v < 8; ++v) {
    const int K2 = (v >> 2) * 16 + half * 8 + 2 * (v & 3);
    B0.d[v] = *(const unsigned*)(BsK + (cb + m) * BSTRB + K2);
    B1.d[v] = *(const unsigned*)(BsK + (cb + 16 + m) * BSTRB + K2);
  }
  c00 = __builtin_amdgcn_wmma_f32_16x16x32_bf16(false, A0.v, false, B0.v, (short)0, c00, false, false);
  c01 = __builtin_amdgcn_wmma_f32_16x16x32_bf16(false, A0.v, false, B1.v, (short)0, c01, false, false);
  c10 = __builtin_amdgcn_wmma_f32_16x16x32_bf16(false, A1.v, false, B0.v, (short)0, c10, false, false);
  c11 = __builtin_amdgcn_wmma_f32_16x16x32_bf16(false, A1.v, false, B1.v, (short)0, c11, false, false);
}

__global__ __launch_bounds__(256) void gat_attn_kernel(
    const float* __restrict__ adj, const unsigned* __restrict__ WhT,
    const float* __restrict__ e1g, const float* __restrict__ e2g,
    float* __restrict__ out) {
  extern __shared__ char smem[];
  unsigned short* Pl   = (unsigned short*)smem;                 // 32 x 2048 bf16 P chunk
  unsigned short* BsA  = (unsigned short*)(smem + LDS_B_OFF);   // 256 cols x 64 K (buffer A)
  unsigned short* BsB  = (unsigned short*)(smem + LDS_B_OFF + LDS_B_ONE);
  float*          red2 = (float*)(smem + LDS_R2_OFF);           // 32 rows x 8 wave partials
  float*          mxs  = (float*)(smem + LDS_MX_OFF);
  float*          invs = (float*)(smem + LDS_INV_OFF);
  const unsigned bsA = (unsigned)(size_t)(void*)BsA;            // LDS byte offsets for async dst
  const unsigned bsB = (unsigned)(size_t)(void*)BsB;

  const int i0 = blockIdx.x * MROWS;
  const int t  = threadIdx.x;
  const int jb = t * 32;
  const int lane = t & 31, w = t >> 5;
  const int m = lane & 15, half = lane >> 4;

  // ---- pass 1: adj -> mask bits (regs) + masked row max ----
  unsigned maskw[MROWS];
  {
    float e2r[32];
    const float4* ep = (const float4*)(e2g + jb);
    #pragma unroll
    for (int q = 0; q < 8; ++q) {
      F4 u; u.v = ep[q];
      e2r[q*4+0] = u.f[0]; e2r[q*4+1] = u.f[1];
      e2r[q*4+2] = u.f[2]; e2r[q*4+3] = u.f[3];
    }
    #pragma unroll
    for (int i = 0; i < MROWS; ++i) {
      const float e1v = e1g[i0 + i];   // uniform -> scalar load
      const float4* ar = (const float4*)(adj + (size_t)(i0 + i) * N_NODES + jb);
      unsigned mw = 0u;
      float mloc = -3.0e38f;
      #pragma unroll
      for (int q = 0; q < 8; ++q) {
        F4 av; av.v = ar[q];
        #pragma unroll
        for (int s2 = 0; s2 < 4; ++s2) {
          const int qq = q * 4 + s2;
          const bool on = (av.f[s2] > 0.f) || ((jb + qq) == (i0 + i)); // adj + I self-loop
          float e = e1v + e2r[qq];
          e = (e > 0.f) ? e : (ALPHA * e);
          if (on) { mw |= (1u << qq); mloc = fmaxf(mloc, e); }
        }
      }
      maskw[i] = mw;
      const float wm = wave_max(mloc);
      if (lane == 0) red2[i * 8 + w] = wm;
    }
  }
  __syncthreads();
  if (t < MROWS) {
    float mxv = -3.0e38f;
    #pragma unroll
    for (int w2 = 0; w2 < 8; ++w2) mxv = fmaxf(mxv, red2[t * 8 + w2]);
    mxs[t] = mxv;
  }

  f32x8 c00 = {}, c01 = {}, c10 = {}, c11 = {};

  #pragma unroll
  for (int c = 0; c < NCHUNK; ++c) {
    __syncthreads();  // P buffer free; mxs visible

    // ---- write bf16 P chunk (this thread's 8-j sub-slice) + wave row sums ----
    {
      float e2r[32];
      const float4* ep = (const float4*)(e2g + jb);
      #pragma unroll
      for (int q = 0; q < 8; ++q) {
        F4 u; u.v = ep[q];
        e2r[q*4+0] = u.f[0]; e2r[q*4+1] = u.f[1];
        e2r[q*4+2] = u.f[2]; e2r[q*4+3] = u.f[3];
      }
      #pragma unroll
      for (int i = 0; i < MROWS; ++i) {
        const unsigned mw = maskw[i];
        const float e1v = e1g[i0 + i];
        const float mxi = mxs[i];
        float ssum = 0.f;
        #pragma unroll
        for (int q = 0; q < 8; q += 2) {
          const int b = 8 * c + q;
          float ea = e1v + e2r[b];     ea = (ea > 0.f) ? ea : (ALPHA * ea);
          float eb = e1v + e2r[b + 1]; eb = (eb > 0.f) ? eb : (ALPHA * eb);
          const float pa = ((mw >> b) & 1u)       ? __expf(ea - mxi) : 0.f;
          const float pb = ((mw >> (b + 1)) & 1u) ? __expf(eb - mxi) : 0.f;
          ssum += pa + pb;
          const unsigned pk = (unsigned)f2bf(pa) | ((unsigned)f2bf(pb) << 16);
          *(unsigned*)(Pl + i * PSTR + t * 8 + q) = pk;
        }
        const float ws = wave_sum(ssum);
        if (lane == 0) {
          if (c == 0) red2[i * 8 + w] = ws;
          else        red2[i * 8 + w] += ws;
        }
      }
    }

    if (c == NCHUNK - 1) {  // finish softmax denominators
      __syncthreads();
      if (t < MROWS) {
        float tot = 0.f;
        #pragma unroll
        for (int w2 = 0; w2 < 8; ++w2) tot += red2[t * 8 + w2];
        invs[t] = 1.0f / tot;   // visible after first barrier inside k-loop
      }
    }

    // ---- k-loop: out += P @ Wh ; B tiles via async DMA, ping-pong LDS buffers ----
    issue_tile_async(WhT, t, 0, c, bsA);
    #pragma unroll 1
    for (int k0 = 0; k0 < CHUNK; k0 += 128) {
      asm volatile("s_wait_asynccnt 0x0" ::: "memory");
      __syncthreads();                                  // buffer A ready everywhere
      issue_tile_async(WhT, t, k0 + 64, c, bsB);        // overlaps with mma below
      mma_step(Pl, BsA,      k0,      m, half, w, c00, c01, c10, c11);
      mma_step(Pl, BsA + 32, k0 + 32, m, half, w, c00, c01, c10, c11);

      asm volatile("s_wait_asynccnt 0x0" ::: "memory");
      __syncthreads();                                  // buffer B ready everywhere
      if (k0 + 128 < CHUNK) issue_tile_async(WhT, t, k0 + 128, c, bsA);
      mma_step(Pl, BsB,      k0 + 64, m, half, w, c00, c01, c10, c11);
      mma_step(Pl, BsB + 32, k0 + 96, m, half, w, c00, c01, c10, c11);
    }
  }

  // ---- epilogue: apply 1/l, store (C layout: VGPR r -> M = r + 8*half, N = lane&15) ----
  #pragma unroll
  for (int r = 0; r < 8; ++r) {
    const int M = r + 8 * half;
    const float sc0 = invs[M];
    const float sc1 = invs[M + 16];
    out[(size_t)(i0 + M) * FDIM + w * 32 + m]           = c00[r] * sc0;
    out[(size_t)(i0 + M) * FDIM + w * 32 + 16 + m]      = c01[r] * sc0;
    out[(size_t)(i0 + 16 + M) * FDIM + w * 32 + m]      = c10[r] * sc1;
    out[(size_t)(i0 + 16 + M) * FDIM + w * 32 + 16 + m] = c11[r] * sc1;
  }
}

// ============================================================
extern "C" void kernel_launch(void* const* d_in, const int* in_sizes, int n_in,
                              void* d_out, int out_size, void* d_ws, size_t ws_size,
                              hipStream_t stream) {
  (void)in_sizes; (void)n_in; (void)out_size; (void)ws_size;
  const float* adj = (const float*)d_in[0];
  const float* x   = (const float*)d_in[1];
  const float* W   = (const float*)d_in[2];
  const float* a   = (const float*)d_in[3];
  float* out = (float*)d_out;

  unsigned short* whbf = (unsigned short*)d_ws;                           // 4 MB
  unsigned* WhT = (unsigned*)((char*)d_ws + (size_t)N_NODES * FDIM * 2);  // 4 MB packed-pair transpose
  float* e1 = (float*)((char*)d_ws + (size_t)N_NODES * FDIM * 6);         // 32 KB
  float* e2 = e1 + N_NODES;                                               // 32 KB

  gat_wh_kernel<<<N_NODES, 256, 0, stream>>>(x, W, a, whbf, e1, e2);
  gat_tr_kernel<<<256, 256, 0, stream>>>(whbf, WhT);

  hipFuncSetAttribute((const void*)gat_attn_kernel,
                      hipFuncAttributeMaxDynamicSharedMemorySize, LDS_TOTAL);
  gat_attn_kernel<<<N_NODES / MROWS, 256, LDS_TOTAL, stream>>>(adj, WhT, e1, e2, out);
}
